// GCNModelVAE_31121333027041
// MI455X (gfx1250) — matrix operand order repaired and checked by reading, not compiled, not run on hardware
//
#include <hip/hip_runtime.h>

// GCN-VAE forward for MI455X (gfx1250, wave32, WMMA).
// Pipeline: bf16-convert -> WMMA GEMM (x@W1) -> edge-scatter SpMM -> relu
//           -> WMMA GEMMs (h1@W2, h1@W3) -> SpMM -> z=mu -> WMMA GEMM z@z^T.
// Dominant cost: 256 MB adj_recon write (HBM-bound, ~11us at 23.3 TB/s).
// All GEMMs use v_wmma_f32_16x16x32_bf16, fp32 accumulate, 2x4 register
// blocking per wave (8 independent accumulator chains to hide XDL latency),
// and the decoder output is stored non-temporally to avoid flushing L2.

typedef __bf16 bf16_t;
typedef __attribute__((ext_vector_type(16))) __bf16 v16bf;
typedef __attribute__((ext_vector_type(8)))  __bf16 v8bf;
typedef __attribute__((ext_vector_type(8)))  float  v8f;

#define N_NODES 8192
#define F_IN    512
#define H1DIM   256
#define H2DIM   64
#define E_EDGES 262144

// ---------------- elementwise helpers ----------------

__global__ void k_f32_to_bf16(const float* __restrict__ src, bf16_t* __restrict__ dst, int n) {
  int i = blockIdx.x * blockDim.x + threadIdx.x;
  if (i < n) dst[i] = (bf16_t)src[i];
}

// W[K,Nout] fp32 -> Wt[Nout,K] bf16 (so GEMM B-fragment loads are contiguous)
__global__ void k_transpose_to_bf16(const float* __restrict__ W, bf16_t* __restrict__ Wt,
                                    int K, int Nout) {
  int i = blockIdx.x * blockDim.x + threadIdx.x;
  if (i < K * Nout) {
    int k = i / Nout, n = i % Nout;
    Wt[n * K + k] = (bf16_t)W[i];
  }
}

__global__ void k_zero(float* __restrict__ p, int n) {
  int i = blockIdx.x * blockDim.x + threadIdx.x;
  if (i < n) p[i] = 0.0f;
}

__global__ void k_relu_to_bf16(const float* __restrict__ src, bf16_t* __restrict__ dst, int n) {
  int i = blockIdx.x * blockDim.x + threadIdx.x;
  if (i < n) { float v = src[i]; dst[i] = (bf16_t)(v > 0.0f ? v : 0.0f); }
}

// z = mu (copy fp32) and also produce bf16 z for the final WMMA GEMM
__global__ void k_copy_and_bf16(const float* __restrict__ src, float* __restrict__ dstF,
                                bf16_t* __restrict__ dstB, int n) {
  int i = blockIdx.x * blockDim.x + threadIdx.x;
  if (i < n) { float v = src[i]; dstF[i] = v; dstB[i] = (bf16_t)v; }
}

// ---------------- SpMM: dst[rows[e],:] += vals[e] * src[cols[e],:] ----------------
// One block per edge, one thread per feature; coalesced gather + f32 atomic scatter.
__global__ void k_spmm_scatter(const int* __restrict__ rows, const int* __restrict__ cols,
                               const float* __restrict__ vals,
                               const float* __restrict__ src, float* __restrict__ dst, int D) {
  int e = blockIdx.x;
  int f = threadIdx.x;
  int r = rows[e];
  int c = cols[e];
  float v = vals[e];
  atomicAdd(&dst[(size_t)r * D + f], v * src[(size_t)c * D + f]);
}

// ---------------- WMMA bf16 GEMM:  C[M,N] = A[M,K] * Bt[N,K]^T ----------------
// One wave computes a (16*BM) x (16*BN) macro-tile: BM*BN independent
// accumulators so consecutive v_wmma ops have no RAW hazard, and A/B
// fragment loads are amortized BNx / BMx. 256-thread blocks = 8 waves.
// Fragment layouts per CDNA5 ISA 7.12.2 (wave32):
//  A 16x32 bf16: lane half 0 -> K=k0+0..7 / k0+16..23 ; half 1 -> K=k0+8..15 / k0+24..31
//  B 32x16 bf16: col = lane&15; half 0 -> K=k0+0..7 / k0+16..23 ; half 1 -> K=k0+8..15 / k0+24..31 (same as A)
//  C/D 16x16 f32: row = vgpr + 8*(lane>=16), col = lane&15
template <int BM, int BN, bool NT>
__global__ void __launch_bounds__(256)
k_gemm_bf16_blk(const bf16_t* __restrict__ A, const bf16_t* __restrict__ Bt,
                float* __restrict__ C, int M, int N, int K) {
  const int lane = threadIdx.x & 31;
  const int wave = threadIdx.x >> 5;
  const int mtN = (N >> 4) / BN;                 // macro-tiles along N
  const int mtM = (M >> 4) / BM;                 // macro-tiles along M
  const int t = blockIdx.x * 8 + wave;
  if (t >= mtM * mtN) return;                    // wave-uniform guard (EXEC all-1s)
  const int tm0 = (t / mtN) * BM;                // first 16-tile row index
  const int tn0 = (t % mtN) * BN;                // first 16-tile col index
  const int half = lane >> 4;                    // 0: lanes 0-15, 1: lanes 16-31
  const int l16  = lane & 15;

  const bf16_t* arow[BM];
#pragma unroll
  for (int i = 0; i < BM; ++i)
    arow[i] = A + (size_t)((tm0 + i) * 16 + l16) * K;
  const bf16_t* brow[BN];
#pragma unroll
  for (int j = 0; j < BN; ++j)
    brow[j] = Bt + (size_t)((tn0 + j) * 16 + l16) * K;

  v8f acc[BM][BN] = {};

  for (int k0 = 0; k0 < K; k0 += 32) {
    const int ka = k0 + half * 8;                // A half-lane K base
    const int kb = k0 + half * 8;                // B half-lane K base
    v16bf av[BM], bv[BN];
#pragma unroll
    for (int i = 0; i < BM; ++i) {
      v8bf a0 = *(const v8bf*)(arow[i] + ka);        // K = ka..ka+7
      v8bf a1 = *(const v8bf*)(arow[i] + ka + 16);   // K = ka+16..ka+23
#pragma unroll
      for (int e = 0; e < 8; ++e) { av[i][e] = a0[e]; av[i][e + 8] = a1[e]; }
    }
#pragma unroll
    for (int j = 0; j < BN; ++j) {
      v8bf b0 = *(const v8bf*)(brow[j] + kb);        // K = kb..kb+7 (col l16)
      v8bf b1 = *(const v8bf*)(brow[j] + kb + 16);   // K = kb+16..kb+23
#pragma unroll
      for (int e = 0; e < 8; ++e) { bv[j][e] = b0[e]; bv[j][e + 8] = b1[e]; }
    }
#pragma unroll
    for (int i = 0; i < BM; ++i)
#pragma unroll
      for (int j = 0; j < BN; ++j)
        acc[i][j] = __builtin_amdgcn_wmma_f32_16x16x32_bf16(
            /*neg_a=*/false, av[i], /*neg_b=*/false, bv[j],
            /*c_mod=*/(short)0, acc[i][j], /*reuse_a=*/false, /*reuse_b=*/false);
  }

#pragma unroll
  for (int i = 0; i < BM; ++i) {
#pragma unroll
    for (int j = 0; j < BN; ++j) {
      float* cptr = C + (size_t)((tm0 + i) * 16 + half * 8) * N
                      + (size_t)(tn0 + j) * 16 + l16;
#pragma unroll
      for (int v = 0; v < 8; ++v) {
        if (NT) __builtin_nontemporal_store(acc[i][j][v], cptr + (size_t)v * N);
        else    cptr[(size_t)v * N] = acc[i][j][v];
      }
    }
  }
}

// ---------------- driver ----------------

static inline size_t align256(size_t x) { return (x + 255) & ~(size_t)255; }

extern "C" void kernel_launch(void* const* d_in, const int* in_sizes, int n_in,
                              void* d_out, int out_size, void* d_ws, size_t ws_size,
                              hipStream_t stream) {
  const float* x        = (const float*)d_in[0];   // [8192, 512]
  const int*   adj_rows = (const int*)  d_in[1];   // [E]
  const int*   adj_cols = (const int*)  d_in[2];   // [E]
  const float* adj_vals = (const float*)d_in[3];   // [E]
  const float* W1       = (const float*)d_in[4];   // [512, 256]
  const float* W2       = (const float*)d_in[5];   // [256, 64]
  const float* W3       = (const float*)d_in[6];   // [256, 64]

  // Output layout: adj_recon | z | mu | logvar (fp32, concatenated flat)
  float* out_adj    = (float*)d_out;
  float* out_z      = out_adj + (size_t)N_NODES * N_NODES;
  float* out_mu     = out_z   + (size_t)N_NODES * H2DIM;
  float* out_logvar = out_mu  + (size_t)N_NODES * H2DIM;

  // Workspace carve-out (~35 MB), 256B-aligned regions
  char* w = (char*)d_ws;
  size_t off = 0;
  bf16_t* xb  = (bf16_t*)(w + off); off = align256(off + (size_t)N_NODES * F_IN  * 2);
  bf16_t* W1t = (bf16_t*)(w + off); off = align256(off + (size_t)H1DIM  * F_IN  * 2);
  bf16_t* W2t = (bf16_t*)(w + off); off = align256(off + (size_t)H2DIM  * H1DIM * 2);
  bf16_t* W3t = (bf16_t*)(w + off); off = align256(off + (size_t)H2DIM  * H1DIM * 2);
  float*  XW1 = (float*) (w + off); off = align256(off + (size_t)N_NODES * H1DIM * 4);
  float*  h1  = (float*) (w + off); off = align256(off + (size_t)N_NODES * H1DIM * 4);
  bf16_t* h1b = (bf16_t*)(w + off); off = align256(off + (size_t)N_NODES * H1DIM * 2);
  float*  HW2 = (float*) (w + off); off = align256(off + (size_t)N_NODES * H2DIM * 4);
  float*  HW3 = (float*) (w + off); off = align256(off + (size_t)N_NODES * H2DIM * 4);
  bf16_t* zb  = (bf16_t*)(w + off); off = align256(off + (size_t)N_NODES * H2DIM * 2);
  (void)ws_size; (void)in_sizes; (void)n_in; (void)out_size;

  const int TB = 256;
  auto grid1d = [](int n, int tb) { return (n + tb - 1) / tb; };
  // macro-tile grid: (M/32 rows) x (N/64 cols) waves, 8 waves per block
  auto gemm_blocks = [](int M, int N) { return ((M / 32) * (N / 64)) / 8; };

  // 1) bf16 conversions / weight transposes
  k_f32_to_bf16<<<grid1d(N_NODES * F_IN, TB), TB, 0, stream>>>(x, xb, N_NODES * F_IN);
  k_transpose_to_bf16<<<grid1d(F_IN * H1DIM, TB), TB, 0, stream>>>(W1, W1t, F_IN, H1DIM);
  k_transpose_to_bf16<<<grid1d(H1DIM * H2DIM, TB), TB, 0, stream>>>(W2, W2t, H1DIM, H2DIM);
  k_transpose_to_bf16<<<grid1d(H1DIM * H2DIM, TB), TB, 0, stream>>>(W3, W3t, H1DIM, H2DIM);

  // 2) XW1 = x @ W1   (8192x256, K=512): 1024 macro-tiles -> 128 blocks
  k_gemm_bf16_blk<2, 4, false><<<gemm_blocks(N_NODES, H1DIM), TB, 0, stream>>>(
      xb, W1t, XW1, N_NODES, H1DIM, F_IN);

  // 3) SpMM1: h1 = A @ XW1 (scatter), then relu -> bf16
  k_zero<<<grid1d(N_NODES * H1DIM, TB), TB, 0, stream>>>(h1, N_NODES * H1DIM);
  k_spmm_scatter<<<E_EDGES, H1DIM, 0, stream>>>(adj_rows, adj_cols, adj_vals, XW1, h1, H1DIM);
  k_relu_to_bf16<<<grid1d(N_NODES * H1DIM, TB), TB, 0, stream>>>(h1, h1b, N_NODES * H1DIM);

  // 4) HW2 = h1 @ W2, HW3 = h1 @ W3   (8192x64, K=256): 256 macro-tiles -> 32 blocks each
  k_gemm_bf16_blk<2, 4, false><<<gemm_blocks(N_NODES, H2DIM), TB, 0, stream>>>(
      h1b, W2t, HW2, N_NODES, H2DIM, H1DIM);
  k_gemm_bf16_blk<2, 4, false><<<gemm_blocks(N_NODES, H2DIM), TB, 0, stream>>>(
      h1b, W3t, HW3, N_NODES, H2DIM, H1DIM);

  // 5) SpMM2/3: mu = A @ HW2, logvar = A @ HW3 (directly into d_out)
  k_zero<<<grid1d(N_NODES * H2DIM, TB), TB, 0, stream>>>(out_mu, N_NODES * H2DIM);
  k_zero<<<grid1d(N_NODES * H2DIM, TB), TB, 0, stream>>>(out_logvar, N_NODES * H2DIM);
  k_spmm_scatter<<<E_EDGES, H2DIM, 0, stream>>>(adj_rows, adj_cols, adj_vals, HW2, out_mu, H2DIM);
  k_spmm_scatter<<<E_EDGES, H2DIM, 0, stream>>>(adj_rows, adj_cols, adj_vals, HW3, out_logvar, H2DIM);

  // 6) z = mu (copy) + bf16 version for the decoder GEMM
  k_copy_and_bf16<<<grid1d(N_NODES * H2DIM, TB), TB, 0, stream>>>(
      out_mu, out_z, zb, N_NODES * H2DIM);

  // 7) adj_recon = z @ z^T  (8192x8192, K=64). Bt == z itself. 32768 macro-tiles
  //    -> 4096 blocks; non-temporal stores keep the 256 MB stream out of L2.
  k_gemm_bf16_blk<2, 4, true><<<gemm_blocks(N_NODES, N_NODES), TB, 0, stream>>>(
      zb, zb, out_adj, N_NODES, N_NODES, H2DIM);
}
